// GNNPlusModel_28372553957732
// MI455X (gfx1250) — compile-verified, matrix operand order checked by reading, not run on hardware
//
#include <hip/hip_runtime.h>
#include <hip/hip_bf16.h>

// ---------------------------------------------------------------------------
// GNNPlus (GINE + FFN) forward for MI455X / gfx1250, wave32 + WMMA.
// Dense MLPs -> v_wmma_f32_16x16x32_f16 (fp32 accumulate), 16x64 strip/wave.
// Edge gather/scatter-add -> bandwidth-bound VALU kernel with f32 atomics.
// ---------------------------------------------------------------------------

typedef __attribute__((ext_vector_type(16))) _Float16 v16h;
typedef __attribute__((ext_vector_type(8)))  float    v8f;

#define NNODES 50000
#define NEDGES 800000
#define FNODE  64
#define PEDIM  16
#define FEDGE  16
#define HDIM   128
#define NLAYER 4
#define NGRAPH 64
#define NCLASS 10
#define KIN    96   // 80 padded to multiple of 32

// ---------------------------------------------------------------------------
// Weight pre-pack: f32 row-major [Ksrc,Nsrc] -> f16 WMMA-B tile layout.
// Tile (tk,tn) of 32x16 stored as 32 lanes x 16 halfs contiguous:
//   packed[tile][lane][j] = W[k0 + 16*(lane>=16) + j][tn*16 + (lane&15)]
// ---------------------------------------------------------------------------
__global__ void prepack_b(const float* __restrict__ W, _Float16* __restrict__ dst,
                          int Ksrc, int Nsrc, int Kpad, int Npad) {
    int t = blockIdx.x * blockDim.x + threadIdx.x;
    if (t >= Kpad * Npad) return;
    int k = t / Npad;
    int n = t - k * Npad;
    int tk = k >> 5, tn = n >> 4;
    int lane = (n & 15) | (((k >> 4) & 1) << 4);
    int j = k & 15;
    float v = (k < Ksrc && n < Nsrc) ? W[k * Nsrc + n] : 0.0f;
    dst[((size_t)(tk * (Npad >> 4) + tn)) * 512 + lane * 16 + j] = (_Float16)v;
}

// ---------------------------------------------------------------------------
// WMMA GEMM: C[M,ncStore] = act(A[M,K] @ Wp[K,Nc] + bias)
// One wave32 per 16 x (NT*16) output strip: one A fetch/convert feeds NT
// WMMAs. K multiple of 32, M multiple of 16, Nc multiple of NT*16.
// ---------------------------------------------------------------------------
template <int NT>
__global__ __launch_bounds__(128)
void gemm_wmma(const float* __restrict__ A, const _Float16* __restrict__ Wp,
               const float* __restrict__ bias, float* __restrict__ C,
               int M, int K, int Nc, int relu, int ncStore) {
    const int lane    = threadIdx.x & 31;
    const int wave    = threadIdx.x >> 5;
    const int tilesN  = Nc >> 4;          // 16-col tiles
    const int groupsN = tilesN / NT;      // strips of NT tiles
    const int tilesM  = M >> 4;
    const int g = blockIdx.x * 4 + wave;  // wave-uniform
    if (g >= tilesM * groupsN) return;    // wave-uniform: EXEC stays full
    const int tm  = g / groupsN;
    const int tn0 = (g - tm * groupsN) * NT;

    const int r16 = lane & 15;
    const int hi  = lane >> 4;            // 0: K 0..15, 1: K 16..31
    v8f c[NT];
#pragma unroll
    for (int u = 0; u < NT; ++u) c[u] = (v8f){};

    const float*    aBase = A + (size_t)(tm * 16 + r16) * K + hi * 16;
    const _Float16* bBase = Wp + (size_t)tn0 * 512 + lane * 16;
    const int kTiles = K >> 5;

    for (int kt = 0; kt < kTiles; ++kt) {
        const float* ap = aBase + kt * 32;
        if (kt + 1 < kTiles)
            __builtin_prefetch(ap + 32, 0, 1);   // global_prefetch_b8 (A stream)
        v16h a;
#pragma unroll
        for (int j = 0; j < 16; ++j) a[j] = (_Float16)ap[j];
        const _Float16* bp = bBase + (size_t)kt * tilesN * 512;
#pragma unroll
        for (int u = 0; u < NT; ++u) {
            v16h b;
#pragma unroll
            for (int j = 0; j < 16; ++j) b[j] = bp[u * 512 + j];
            // D = A(16x32 f16) x B(32x16 f16) + C (f32)
            c[u] = __builtin_amdgcn_wmma_f32_16x16x32_f16(false, a, false, b,
                                                          (short)0, c[u], false, false);
        }
    }

#pragma unroll
    for (int u = 0; u < NT; ++u) {
        const int col = (tn0 + u) * 16 + (lane & 15);
        if (col < ncStore) {
            float bv = bias ? bias[col] : 0.0f;
#pragma unroll
            for (int r = 0; r < 8; ++r) {
                int row = tm * 16 + r + hi * 8;
                float v = c[u][r] + bv;
                if (relu) v = fmaxf(v, 0.0f);
                C[(size_t)row * ncStore + col] = v;
            }
        }
    }
}

// ---------------------------------------------------------------------------
// Edge stage: msg = relu(h[src] + edge_attr @ We + be); atomic scatter to agg.
// 128 threads per edge (one per channel). Memory-bound; atomics to L2.
// ---------------------------------------------------------------------------
__global__ void edge_msg_agg(const float* __restrict__ h, const float* __restrict__ ea,
                             const int* __restrict__ ei, const float* __restrict__ We,
                             const float* __restrict__ be, float* __restrict__ agg) {
    int t = blockIdx.x * blockDim.x + threadIdx.x;
    int e = t >> 7;
    int c = t & 127;
    if (e >= NEDGES) return;
    int src = ei[e];
    int dst = ei[NEDGES + e];
    float acc = be[c];
    const float* eaRow = ea + (size_t)e * FEDGE;
#pragma unroll
    for (int k = 0; k < FEDGE; ++k) acc = fmaf(eaRow[k], We[k * HDIM + c], acc);
    float msg = fmaxf(h[(size_t)src * HDIM + c] + acc, 0.0f);
    atomicAdd(&agg[(size_t)dst * HDIM + c], msg);
}

// ---------------------------------------------------------------------------
// Elementwise / reduction helpers
// ---------------------------------------------------------------------------
__global__ void zero_f32(float* __restrict__ p, int n) {
    int t = blockIdx.x * blockDim.x + threadIdx.x;
    if (t < n) p[t] = 0.0f;
}

__global__ void concat_xpe(const float* __restrict__ x, const float* __restrict__ pe,
                           float* __restrict__ A) {
    int t = blockIdx.x * blockDim.x + threadIdx.x;
    int n = t / KIN, c = t - n * KIN;
    if (n >= NNODES) return;
    float v = 0.0f;
    if (c < FNODE)              v = x[(size_t)n * FNODE + c];
    else if (c < FNODE + PEDIM) v = pe[(size_t)n * PEDIM + (c - FNODE)];
    A[(size_t)n * KIN + c] = v;
}

__global__ void gin_pre(const float* __restrict__ h, const float* __restrict__ agg,
                        const float* __restrict__ epsArr, int l, float* __restrict__ z0,
                        int n) {
    int t = blockIdx.x * blockDim.x + threadIdx.x;
    if (t >= n) return;
    float e = 1.0f + epsArr[l];
    z0[t] = fmaf(e, h[t], agg[t]);
}

__global__ void add2(const float* __restrict__ a, const float* __restrict__ b,
                     float* __restrict__ o, int n) {
    int t = blockIdx.x * blockDim.x + threadIdx.x;
    if (t < n) o[t] = a[t] + b[t];
}

// Split-column reduction: stats[c]=sum, stats[C+c]=sumsq via few atomics.
__global__ void bn_reduce(const float* __restrict__ X, float* __restrict__ stats,
                          int N, int C, int splits) {
    int t = blockIdx.x * blockDim.x + threadIdx.x;
    if (t >= C * splits) return;
    int c = t % C, s = t / C;
    float sum = 0.0f, sq = 0.0f;
    for (int r = s; r < N; r += splits) {
        float v = X[(size_t)r * C + c];
        sum += v; sq = fmaf(v, v, sq);
    }
    atomicAdd(&stats[c], sum);
    atomicAdd(&stats[C + c], sq);
}

__global__ void bn_apply(const float* __restrict__ X, const float* __restrict__ stats,
                         const float* __restrict__ g, const float* __restrict__ bta,
                         const float* __restrict__ res, float* __restrict__ Y,
                         int N, int C, int relu) {
    int t = blockIdx.x * blockDim.x + threadIdx.x;
    if (t >= N * C) return;
    int c = t & (C - 1);
    float m  = stats[c] / (float)N;
    float vr = stats[C + c] / (float)N - m * m;
    float rs = rsqrtf(vr + 1e-5f);
    float v = (X[t] - m) * rs * g[c] + bta[c];
    if (relu) v = fmaxf(v, 0.0f);
    if (res)  v += res[t];
    Y[t] = v;
}

__global__ void pool_acc(const float* __restrict__ h, const int* __restrict__ batch,
                         float* __restrict__ pooled, float* __restrict__ counts) {
    int t = blockIdx.x * blockDim.x + threadIdx.x;
    int n = t >> 7, c = t & 127;
    if (n >= NNODES) return;
    int gidx = batch[n];
    atomicAdd(&pooled[(size_t)gidx * HDIM + c], h[(size_t)n * HDIM + c]);
    if (c == 0) atomicAdd(&counts[gidx], 1.0f);
}

__global__ void pool_div(const float* __restrict__ pooled, const float* __restrict__ counts,
                         float* __restrict__ hg) {
    int t = blockIdx.x * blockDim.x + threadIdx.x;
    if (t >= NGRAPH * HDIM) return;
    int gidx = t >> 7;
    hg[t] = pooled[t] / fmaxf(counts[gidx], 1.0f);
}

// ---------------------------------------------------------------------------
// Host orchestration
// ---------------------------------------------------------------------------
static inline int cdiv(long long a, long long b) { return (int)((a + b - 1) / b); }

extern "C" void kernel_launch(void* const* d_in, const int* in_sizes, int n_in,
                              void* d_out, int out_size, void* d_ws, size_t ws_size,
                              hipStream_t stream) {
    (void)in_sizes; (void)n_in; (void)out_size; (void)ws_size;
    const float* x        = (const float*)d_in[0];
    const float* pe       = (const float*)d_in[1];
    const int*   ei       = (const int*)  d_in[2];
    const float* eattr    = (const float*)d_in[3];
    const int*   batch    = (const int*)  d_in[4];
    const float* W_in     = (const float*)d_in[5];
    const float* b_in     = (const float*)d_in[6];
    const float* We_edge  = (const float*)d_in[7];
    const float* be_edge  = (const float*)d_in[8];
    const float* eps      = (const float*)d_in[9];
    const float* W1_gin   = (const float*)d_in[10];
    const float* b1_gin   = (const float*)d_in[11];
    const float* W2_gin   = (const float*)d_in[12];
    const float* b2_gin   = (const float*)d_in[13];
    const float* bn1_g    = (const float*)d_in[14];
    const float* bn1_b    = (const float*)d_in[15];
    const float* W1_ffn   = (const float*)d_in[16];
    const float* b1_ffn   = (const float*)d_in[17];
    const float* W2_ffn   = (const float*)d_in[18];
    const float* b2_ffn   = (const float*)d_in[19];
    const float* bn2_g    = (const float*)d_in[20];
    const float* bn2_b    = (const float*)d_in[21];
    const float* W_r1     = (const float*)d_in[22];
    const float* b_r1     = (const float*)d_in[23];
    const float* W_r2     = (const float*)d_in[24];
    const float* b_r2     = (const float*)d_in[25];
    float* out = (float*)d_out;

    // ---- workspace carve-out (256B aligned) ----
    char* ws = (char*)d_ws;
    size_t off = 0;
    auto alloc = [&](size_t bytes) -> char* {
        char* p = ws + off;
        off = (off + bytes + 255) & ~(size_t)255;
        return p;
    };
    const size_t NH  = (size_t)NNODES * HDIM;
    const size_t N2H = (size_t)NNODES * 2 * HDIM;

    _Float16* wpIn  = (_Float16*)alloc((size_t)KIN * HDIM * 2);
    _Float16* wpW1g = (_Float16*)alloc((size_t)NLAYER * HDIM * 2 * HDIM * 2);
    _Float16* wpW2g = (_Float16*)alloc((size_t)NLAYER * 2 * HDIM * HDIM * 2);
    _Float16* wpW1f = (_Float16*)alloc((size_t)NLAYER * HDIM * 2 * HDIM * 2);
    _Float16* wpW2f = (_Float16*)alloc((size_t)NLAYER * 2 * HDIM * HDIM * 2);
    _Float16* wpR1  = (_Float16*)alloc((size_t)HDIM * HDIM * 2);
    _Float16* wpR2  = (_Float16*)alloc((size_t)HDIM * 16 * 2);

    float* H    = (float*)alloc(NH * 4);
    float* AGG  = (float*)alloc(NH * 4);
    float* Z0   = (float*)alloc(NH * 4);
    float* Z    = (float*)alloc(NH * 4);
    float* T1   = (float*)alloc(N2H * 4);            // also reused as [N,96] concat
    float* T2   = (float*)alloc(NH * 4);
    float* STAT = (float*)alloc(2 * HDIM * 4);
    float* POOL = (float*)alloc((size_t)NGRAPH * HDIM * 4);
    float* CNT  = (float*)alloc((size_t)NGRAPH * 4);
    float* HG   = (float*)alloc((size_t)NGRAPH * HDIM * 4);
    float* R1O  = (float*)alloc((size_t)NGRAPH * HDIM * 4);

    const int TB = 256;
    auto gemmBlocks4 = [](int M, int Nc) {            // NT=4 strips
        int waves = (M >> 4) * ((Nc >> 4) / 4);
        return cdiv(waves, 4);
    };

    // ---- 1. pre-pack weights into WMMA B-operand layout (f16) ----
    prepack_b<<<cdiv((size_t)KIN * HDIM, TB), TB, 0, stream>>>(W_in, wpIn, 80, HDIM, KIN, HDIM);
    for (int l = 0; l < NLAYER; ++l) {
        prepack_b<<<cdiv(HDIM * 2 * HDIM, TB), TB, 0, stream>>>(
            W1_gin + (size_t)l * HDIM * 2 * HDIM, wpW1g + (size_t)l * HDIM * 2 * HDIM,
            HDIM, 2 * HDIM, HDIM, 2 * HDIM);
        prepack_b<<<cdiv(2 * HDIM * HDIM, TB), TB, 0, stream>>>(
            W2_gin + (size_t)l * 2 * HDIM * HDIM, wpW2g + (size_t)l * 2 * HDIM * HDIM,
            2 * HDIM, HDIM, 2 * HDIM, HDIM);
        prepack_b<<<cdiv(HDIM * 2 * HDIM, TB), TB, 0, stream>>>(
            W1_ffn + (size_t)l * HDIM * 2 * HDIM, wpW1f + (size_t)l * HDIM * 2 * HDIM,
            HDIM, 2 * HDIM, HDIM, 2 * HDIM);
        prepack_b<<<cdiv(2 * HDIM * HDIM, TB), TB, 0, stream>>>(
            W2_ffn + (size_t)l * 2 * HDIM * HDIM, wpW2f + (size_t)l * 2 * HDIM * HDIM,
            2 * HDIM, HDIM, 2 * HDIM, HDIM);
    }
    prepack_b<<<cdiv(HDIM * HDIM, TB), TB, 0, stream>>>(W_r1, wpR1, HDIM, HDIM, HDIM, HDIM);
    prepack_b<<<cdiv(HDIM * 16, TB), TB, 0, stream>>>(W_r2, wpR2, HDIM, NCLASS, HDIM, 16);

    // ---- 2. input projection: h = relu([x||pe] @ W_in + b_in) ----
    concat_xpe<<<cdiv((size_t)NNODES * KIN, TB), TB, 0, stream>>>(x, pe, T1);
    gemm_wmma<4><<<gemmBlocks4(NNODES, HDIM), 128, 0, stream>>>(
        T1, wpIn, b_in, H, NNODES, KIN, HDIM, 1, HDIM);

    // ---- 3. layers ----
    for (int l = 0; l < NLAYER; ++l) {
        // GINE message + sum-aggregate
        zero_f32<<<cdiv(NH, TB), TB, 0, stream>>>(AGG, (int)NH);
        edge_msg_agg<<<cdiv((size_t)NEDGES * HDIM, TB), TB, 0, stream>>>(
            H, eattr, ei, We_edge + (size_t)l * FEDGE * HDIM,
            be_edge + (size_t)l * HDIM, AGG);
        gin_pre<<<cdiv(NH, TB), TB, 0, stream>>>(H, AGG, eps, l, Z0, (int)NH);

        // GIN MLP: T2 = relu(Z0@W1+b1)@W2 + b2
        gemm_wmma<4><<<gemmBlocks4(NNODES, 2 * HDIM), 128, 0, stream>>>(
            Z0, wpW1g + (size_t)l * HDIM * 2 * HDIM, b1_gin + (size_t)l * 2 * HDIM,
            T1, NNODES, HDIM, 2 * HDIM, 1, 2 * HDIM);
        gemm_wmma<4><<<gemmBlocks4(NNODES, HDIM), 128, 0, stream>>>(
            T1, wpW2g + (size_t)l * 2 * HDIM * HDIM, b2_gin + (size_t)l * HDIM,
            T2, NNODES, 2 * HDIM, HDIM, 0, HDIM);

        // BN1 -> relu -> + h  => Z
        zero_f32<<<1, 256, 0, stream>>>(STAT, 2 * HDIM);
        bn_reduce<<<cdiv(HDIM * 256, TB), TB, 0, stream>>>(T2, STAT, NNODES, HDIM, 256);
        bn_apply<<<cdiv(NH, TB), TB, 0, stream>>>(
            T2, STAT, bn1_g + (size_t)l * HDIM, bn1_b + (size_t)l * HDIM, H, Z,
            NNODES, HDIM, 1);

        // FFN: T2 = relu(Z@W1f+b1f)@W2f + b2f
        gemm_wmma<4><<<gemmBlocks4(NNODES, 2 * HDIM), 128, 0, stream>>>(
            Z, wpW1f + (size_t)l * HDIM * 2 * HDIM, b1_ffn + (size_t)l * 2 * HDIM,
            T1, NNODES, HDIM, 2 * HDIM, 1, 2 * HDIM);
        gemm_wmma<4><<<gemmBlocks4(NNODES, HDIM), 128, 0, stream>>>(
            T1, wpW2f + (size_t)l * 2 * HDIM * HDIM, b2_ffn + (size_t)l * HDIM,
            T2, NNODES, 2 * HDIM, HDIM, 0, HDIM);

        // h = BN2(T2 + Z)
        add2<<<cdiv(NH, TB), TB, 0, stream>>>(T2, Z, Z0, (int)NH);
        zero_f32<<<1, 256, 0, stream>>>(STAT, 2 * HDIM);
        bn_reduce<<<cdiv(HDIM * 256, TB), TB, 0, stream>>>(Z0, STAT, NNODES, HDIM, 256);
        bn_apply<<<cdiv(NH, TB), TB, 0, stream>>>(
            Z0, STAT, bn2_g + (size_t)l * HDIM, bn2_b + (size_t)l * HDIM, nullptr, H,
            NNODES, HDIM, 0);
    }

    // ---- 4. global mean pool ----
    zero_f32<<<cdiv(NGRAPH * HDIM + NGRAPH, TB), TB, 0, stream>>>(POOL, NGRAPH * HDIM + NGRAPH);
    pool_acc<<<cdiv(NH, TB), TB, 0, stream>>>(H, batch, POOL, CNT);
    pool_div<<<cdiv(NGRAPH * HDIM, TB), TB, 0, stream>>>(POOL, CNT, HG);

    // ---- 5. readout MLP -> d_out [64,10] ----
    gemm_wmma<4><<<gemmBlocks4(NGRAPH, HDIM), 128, 0, stream>>>(
        HG, wpR1, b_r1, R1O, NGRAPH, HDIM, HDIM, 1, HDIM);
    {
        int waves = (NGRAPH >> 4) * 1;   // Nc=16 -> NT=1
        gemm_wmma<1><<<cdiv(waves, 4), 128, 0, stream>>>(
            R1O, wpR2, b_r2, out, NGRAPH, HDIM, 16, 0, NCLASS);
    }
}
